// TransformerSelfAttention3D_6700148982597
// MI455X (gfx1250) — compile-verified, matrix operand order checked by reading, not run on hardware
//
#include <hip/hip_runtime.h>
#include <hip/hip_bf16.h>

// ---------------------------------------------------------------------------
// 3D self-attention (B=2, C=256, CQK=32, N=4096), flash-attention style.
//   1) proj_kernel: q (pre-scaled by log2(e)/sqrt(C)), k -> bf16 [B,N,32];
//      v -> bf16 channel-major [B,C,N].
//   2) flashattn_kernel: per 16-query tile, stream 64-key blocks:
//      S = Q K^T (4x v_wmma_f32_16x16x32_bf16, K=32 == CQK exact),
//      exp2-domain online softmax with DPP16 butterfly reductions,
//      O += P V (8x WMMA per wave, 64 channels/wave).
// WMMA operands load as two b128 chunks per lane. K/V stay L2-resident.
// ---------------------------------------------------------------------------

typedef __attribute__((ext_vector_type(16))) __bf16 v16bf;
typedef __attribute__((ext_vector_type(8)))  float  v8f;

#define NB   2
#define CCH  256
#define CQK  32
#define NN   4096
// (1/sqrt(256)) * log2(e): softmax runs in exp2 domain
#define QSCALE (0.0625f * 1.44269504088896341f)

// raw v_exp_f32
#define EXP2F(x) __builtin_amdgcn_exp2f(x)

__device__ __forceinline__ unsigned short f2bf(float f) {
  unsigned int u = __float_as_uint(f);
  u += 0x7FFFu + ((u >> 16) & 1u);     // round-to-nearest-even
  return (unsigned short)(u >> 16);
}

union V16U8 { unsigned int u[8]; uint4 q[2]; v16bf v; };

// DPP16 lane-xor within each 16-lane row via basis {1,2,7,15}:
//   xor1 = quad_perm [1,0,3,2] (0xB1), xor2 = quad_perm [2,3,0,1] (0x4E),
//   xor7 = ROW_HALF_MIRROR (0x141),    xor15 = ROW_MIRROR (0x140).
template <int CTRL>
__device__ __forceinline__ float dppf(float x) {
  return __int_as_float(__builtin_amdgcn_update_dpp(
      0, __float_as_int(x), CTRL, 0xf, 0xf, true));
}
__device__ __forceinline__ float red16_max(float x) {
  x = fmaxf(x, dppf<0xB1>(x));
  x = fmaxf(x, dppf<0x4E>(x));
  x = fmaxf(x, dppf<0x141>(x));
  x = fmaxf(x, dppf<0x140>(x));
  return x;
}
__device__ __forceinline__ float red16_sum(float x) {
  x += dppf<0xB1>(x);
  x += dppf<0x4E>(x);
  x += dppf<0x141>(x);
  x += dppf<0x140>(x);
  return x;
}

// Load one 16-bf16-per-lane WMMA operand: two contiguous 16B chunks.
__device__ __forceinline__ void load_op(V16U8& t, const unsigned int* rowbase,
                                        int h) {
  t.q[0] = *(const uint4*)(rowbase + 4 * h);
  t.q[1] = *(const uint4*)(rowbase + 8 + 4 * h);
}

// ----------------------- projection + bf16 convert -------------------------
__global__ void __launch_bounds__(256)
proj_kernel(const float* __restrict__ x,
            const float* __restrict__ Wq, const float* __restrict__ bq,
            const float* __restrict__ Wk, const float* __restrict__ bk,
            const float* __restrict__ Wv, const float* __restrict__ bv,
            unsigned short* __restrict__ qb,   // [B,N,32] (scaled)
            unsigned short* __restrict__ kb,   // [B,N,32]
            unsigned short* __restrict__ vb)   // [B,C,N]  (channel-major!)
{
  __shared__ float xcol[CCH];
  const int tid = threadIdx.x;
  const int n = blockIdx.x & (NN - 1);
  const int b = blockIdx.x >> 12;
  xcol[tid] = x[((b * CCH + tid) * NN) + n];
  __syncthreads();

  float acc = bv[tid];
  const float* wr = Wv + tid * CCH;
  #pragma unroll 8
  for (int c = 0; c < CCH; ++c) acc += wr[c] * xcol[c];
  vb[(b * CCH + tid) * NN + n] = f2bf(acc);

  if (tid < 2 * CQK) {
    const int o = tid & (CQK - 1);
    const bool isq = tid < CQK;
    const float* wm = (isq ? Wq : Wk) + o * CCH;
    float a = isq ? bq[o] : bk[o];
    #pragma unroll 8
    for (int c = 0; c < CCH; ++c) a += wm[c] * xcol[c];
    if (isq) qb[(b * NN + n) * CQK + o] = f2bf(a * QSCALE);
    else     kb[(b * NN + n) * CQK + o] = f2bf(a);
  }
}

// --------------------------- flash attention -------------------------------
// grid: B * N/16 blocks, 128 threads (4 waves); wave w owns V channels
// [64w, 64w+64). Each wave redundantly computes S for its 16-query tile.
__global__ void __launch_bounds__(128)
flashattn_kernel(const float* __restrict__ x,
                 const unsigned short* __restrict__ qb,
                 const unsigned short* __restrict__ kb,
                 const unsigned short* __restrict__ vb,
                 const float* __restrict__ gamma,
                 float* __restrict__ out)
{
  __shared__ __align__(16) unsigned short Pbuf[4][16 * 64];  // per-wave P tile

  const int lane = threadIdx.x & 31;
  const int wave = threadIdx.x >> 5;
  const int qt = blockIdx.x & 255;
  const int b  = blockIdx.x >> 8;
  const int m0 = qt * 16;

  const int ncol = lane & 15;  // column / row-within-16 selector
  const int h    = lane >> 4;  // half-wave selector

  const unsigned int* qb32 = (const unsigned int*)qb;
  const unsigned int* kb32 = (const unsigned int*)kb;
  const unsigned int* vb32 = (const unsigned int*)vb;

  // Q tile in A layout (row = m0 + lane%16)
  V16U8 aq;
  load_op(aq, qb32 + (unsigned)(b * NN + m0 + ncol) * (CQK / 2), h);

  v8f Ov[4];
  float mrun[8], lrun[8];
  #pragma unroll
  for (int t = 0; t < 4; ++t) Ov[t] = (v8f){};
  #pragma unroll
  for (int j = 0; j < 8; ++j) { mrun[j] = -1e30f; lrun[j] = 0.0f; }

  const int c0 = wave * 64;
  const v8f zero = {};
  unsigned short* Pw = Pbuf[wave];
  const unsigned int* Pw32 = (const unsigned int*)Pw;

  for (int k0 = 0; k0 < NN; k0 += 64) {
    // ---- K block: four B matrices (keys k0+16u .. k0+16u+15)
    const unsigned int* kr = kb32 + (unsigned)(b * NN + k0 + ncol) * (CQK / 2);
    V16U8 bk0, bk1, bk2, bk3;
    load_op(bk0, kr,                     h);
    load_op(bk1, kr + 16 * (CQK / 2),    h);
    load_op(bk2, kr + 32 * (CQK / 2),    h);
    load_op(bk3, kr + 48 * (CQK / 2),    h);

    if (k0 + 64 < NN)  // next key block toward L0
      __builtin_prefetch((const void*)(kr + 64 * (CQK / 2)), 0, 0);

    // ---- S = Q K^T (scale + log2e folded into q)
    v8f s0 = __builtin_amdgcn_wmma_f32_16x16x32_bf16(
        false, aq.v, false, bk0.v, (short)0, zero, false, false);
    v8f s1 = __builtin_amdgcn_wmma_f32_16x16x32_bf16(
        false, aq.v, false, bk1.v, (short)0, zero, false, false);
    v8f s2 = __builtin_amdgcn_wmma_f32_16x16x32_bf16(
        false, aq.v, false, bk2.v, (short)0, zero, false, false);
    v8f s3 = __builtin_amdgcn_wmma_f32_16x16x32_bf16(
        false, aq.v, false, bk3.v, (short)0, zero, false, false);

    // ---- exp2-domain online softmax; rows 8h+j across each 16-lane DPP row
    #pragma unroll
    for (int j = 0; j < 8; ++j) {
      float a0 = s0[j], a1 = s1[j], a2 = s2[j], a3 = s3[j];
      float mj = red16_max(fmaxf(fmaxf(a0, a1), fmaxf(a2, a3)));
      float mnew = fmaxf(mrun[j], mj);
      float corr = EXP2F(mrun[j] - mnew);
      float p0 = EXP2F(a0 - mnew);
      float p1 = EXP2F(a1 - mnew);
      float p2 = EXP2F(a2 - mnew);
      float p3 = EXP2F(a3 - mnew);
      float ps = red16_sum((p0 + p1) + (p2 + p3));
      lrun[j] = lrun[j] * corr + ps;
      mrun[j] = mnew;
      Ov[0][j] *= corr; Ov[1][j] *= corr; Ov[2][j] *= corr; Ov[3][j] *= corr;
      const int row = 8 * h + j;
      Pw[row * 64 + ncol]      = f2bf(p0);
      Pw[row * 64 + 16 + ncol] = f2bf(p1);
      Pw[row * 64 + 32 + ncol] = f2bf(p2);
      Pw[row * 64 + 48 + ncol] = f2bf(p3);
    }

    // D-layout -> A-layout transpose through LDS (intra-wave only)
    asm volatile("s_wait_dscnt 0" ::: "memory");
    V16U8 ap0, ap1;   // P as two 16x32 A operands (keys 0..31, 32..63)
    ap0.q[0] = *(const uint4*)(Pw32 + ncol * 32 + 4 * h);
    ap0.q[1] = *(const uint4*)(Pw32 + ncol * 32 + 8 + 4 * h);
    ap1.q[0] = *(const uint4*)(Pw32 + ncol * 32 + 16 + 4 * h);
    ap1.q[1] = *(const uint4*)(Pw32 + ncol * 32 + 24 + 4 * h);

    // ---- O += P V  (V channel-major: consecutive keys contiguous)
    #pragma unroll
    for (int t = 0; t < 4; ++t) {
      const int c = c0 + t * 16 + ncol;
      const unsigned int* vr = vb32 + (((unsigned)(b * CCH + c) * NN + k0) >> 1);
      V16U8 bv0, bv1;
      load_op(bv0, vr,      h);
      load_op(bv1, vr + 16, h);
      if (k0 + 64 < NN)
        __builtin_prefetch((const void*)(vr + 32), 0, 0);
      Ov[t] = __builtin_amdgcn_wmma_f32_16x16x32_bf16(
          false, ap0.v, false, bv0.v, (short)0, Ov[t], false, false);
      Ov[t] = __builtin_amdgcn_wmma_f32_16x16x32_bf16(
          false, ap1.v, false, bv1.v, (short)0, Ov[t], false, false);
    }
  }

  // ---- epilogue: out = x + gamma * (O / l); 8 consecutive n per (lane,t)
  const float g = gamma[0];
  float rinv[8];
  #pragma unroll
  for (int j = 0; j < 8; ++j) rinv[j] = __frcp_rn(lrun[j]);

  #pragma unroll
  for (int t = 0; t < 4; ++t) {
    const int c = c0 + t * 16 + ncol;
    const size_t base = (size_t)(b * CCH + c) * NN + m0 + 8 * h;
    float4 xa = *(const float4*)(x + base);
    float4 xb = *(const float4*)(x + base + 4);
    float4 ra, rb;
    ra.x = xa.x + g * (Ov[t][0] * rinv[0]);
    ra.y = xa.y + g * (Ov[t][1] * rinv[1]);
    ra.z = xa.z + g * (Ov[t][2] * rinv[2]);
    ra.w = xa.w + g * (Ov[t][3] * rinv[3]);
    rb.x = xb.x + g * (Ov[t][4] * rinv[4]);
    rb.y = xb.y + g * (Ov[t][5] * rinv[5]);
    rb.z = xb.z + g * (Ov[t][6] * rinv[6]);
    rb.w = xb.w + g * (Ov[t][7] * rinv[7]);
    *(float4*)(out + base)     = ra;
    *(float4*)(out + base + 4) = rb;
  }
}

// ------------------------------- launcher ----------------------------------
extern "C" void kernel_launch(void* const* d_in, const int* in_sizes, int n_in,
                              void* d_out, int out_size, void* d_ws, size_t ws_size,
                              hipStream_t stream) {
  const float* x     = (const float*)d_in[0];
  const float* Wq    = (const float*)d_in[1];
  const float* bq    = (const float*)d_in[2];
  const float* Wk    = (const float*)d_in[3];
  const float* bk    = (const float*)d_in[4];
  const float* Wv    = (const float*)d_in[5];
  const float* bv    = (const float*)d_in[6];
  const float* gamma = (const float*)d_in[7];
  float* out = (float*)d_out;

  unsigned short* qb = (unsigned short*)d_ws;            // 512 KB
  unsigned short* kb = qb + (size_t)NB * NN * CQK;       // 512 KB
  unsigned short* vb = kb + (size_t)NB * NN * CQK;       // 4 MB
  (void)in_sizes; (void)n_in; (void)out_size; (void)ws_size;

  proj_kernel<<<NB * NN, 256, 0, stream>>>(x, Wq, bq, Wk, bk, Wv, bv, qb, kb, vb);
  flashattn_kernel<<<NB * (NN / 16), 128, 0, stream>>>(x, qb, kb, vb, gamma, out);
}
